// Encoder_82575041233042
// MI455X (gfx1250) — matrix operand order repaired and checked by reading, not compiled, not run on hardware
//
#include <hip/hip_runtime.h>

// ---------------------------------------------------------------------------
// Problem constants (fixed by the reference)
// ---------------------------------------------------------------------------
#define BATCH   2
#define SEQ     2048
#define DMODEL  1024
#define NHEADS  16
#define DK      64
#define DFF     4096
#define NLAYERS 6
#define ROWS    (BATCH * SEQ)          // 4096
#define LN_EPS  1e-5f
#define LOG2E   1.4426950408889634f
#define QK_SCALE 0.125f                // 1/sqrt(64)

typedef __bf16 v16bf __attribute__((ext_vector_type(16)));
typedef __bf16 v8bf  __attribute__((ext_vector_type(8)));
typedef float  v8f   __attribute__((ext_vector_type(8)));

#if defined(__has_builtin)
#if __has_builtin(__builtin_amdgcn_tensor_load_to_lds) && __has_builtin(__builtin_amdgcn_s_wait_tensorcnt)
#define HAVE_TDM 1
#endif
#endif
#ifndef HAVE_TDM
#define HAVE_TDM 0
#endif

// ---------------------------------------------------------------------------
// WMMA helpers (CDNA5 16x16x32 bf16, fp32 accumulate)
// ---------------------------------------------------------------------------
__device__ __forceinline__ v8f wmma_bf16(v16bf a, v16bf b, v8f c) {
    return __builtin_amdgcn_wmma_f32_16x16x32_bf16(
        /*neg_a=*/false, a, /*neg_b=*/false, b,
        /*c_mod=*/(short)0, c, /*reuse_a=*/false, /*reuse_b=*/false);
}

// A-fragment (16 rows x 32 K, 16-bit): lane<16 holds row=lane, K {0..15};
// lane>=16 holds row=lane-16, K {16..31}.  Two b128 loads.
__device__ __forceinline__ v16bf frag_a_ld(const __bf16* tile, int stride, int lane) {
    const int r = lane & 15, h = lane >> 4;
    const __bf16* p = tile + (size_t)r * stride;
    v8bf lo = *(const v8bf*)(p + h * 16);
    v8bf hi = *(const v8bf*)(p + h * 16 + 8);
    return __builtin_shufflevector(lo, hi, 0,1,2,3,4,5,6,7,8,9,10,11,12,13,14,15);
}

// B-fragment (32 K x 16 N) sourced from a transposed tile T[n][k]:
// lane holds column n=lane%16, K chunk (lane/16)*16 .. +15 (contiguous).
__device__ __forceinline__ v16bf frag_b_ld(const __bf16* tileT, int stride, int lane) {
    const int n = lane & 15, h = lane >> 4;
    const __bf16* p = tileT + (size_t)n * stride + h * 16;
    v8bf lo = *(const v8bf*)(p);
    v8bf hi = *(const v8bf*)(p + 8);
    return __builtin_shufflevector(lo, hi, 0,1,2,3,4,5,6,7,8,9,10,11,12,13,14,15);
}

#if HAVE_TDM
// ---------------------------------------------------------------------------
// Tensor Data Mover: DMA a 2D bf16 tile [128 rows x 32 cols] global -> LDS.
// D# per CDNA5 ISA 8.3/8.4.  Padding: after every 64 B (one 32-elem row) add
// 16 B -> LDS row stride 80 B, matching the [..][40] __bf16 layout used by the
// WMMA fragment loaders (bank-conflict-free, 16 B aligned chunks).
//   pad_interval = 3  (2<<3 = 16 DWORDs = 64 B before pad)
//   pad_amount   = 3  (4 DWORDs = 16 B of pad)
// Tensor dims set huge so the OOB clamp never triggers (tiles are in-bounds).
// ---------------------------------------------------------------------------
typedef unsigned int tdm_v4u __attribute__((ext_vector_type(4)));
typedef int          tdm_v4i __attribute__((ext_vector_type(4)));
typedef int          tdm_v8i __attribute__((ext_vector_type(8)));

__device__ __forceinline__ void tdm_load_tile(unsigned lds_addr, const __bf16* g,
                                              int stride_elems) {
    const unsigned long long ga = (unsigned long long)(size_t)g;
    tdm_v4u g0 = { 1u,                                   // count=1, user mode
                   lds_addr,                             // LDS byte address
                   (unsigned)ga,                         // global addr [31:0]
                   (unsigned)((ga >> 32) & 0x01FFFFFFu)  // global addr [56:32]
                     | 0x80000000u };                    // type=2 ("image")
    tdm_v8i g1 = { (int)((1u << 16)     // data_size = 1 -> 2 bytes (bf16)
                       | (1u << 20)     // pad_enable
                       | (3u << 22)     // pad_interval: 16 DWORDs
                       | (3u << 25)),   // pad_amount:   4 DWORDs
                   0,                                    // atomic barrier addr
                   0x4000,                               // tensor_dim0 = 1<<30
                   (int)(0x4000u | (32u << 16)),         // tensor_dim1 = 1<<30, tile_dim0=32
                   128,                                  // tile_dim1=128, tile_dim2=0
                   stride_elems,                         // tensor_dim0_stride (elems)
                   0, 0 };
    tdm_v4i z4 = {0, 0, 0, 0};
#if __clang_major__ >= 23
    tdm_v8i z8 = {0, 0, 0, 0, 0, 0, 0, 0};
    __builtin_amdgcn_tensor_load_to_lds(g0, g1, z4, z4, z8, 0);
#else
    __builtin_amdgcn_tensor_load_to_lds(g0, g1, z4, z4, 0);
#endif
}
#endif // HAVE_TDM

// ---------------------------------------------------------------------------
// Weight convert + transpose: in fp32 [K][N] -> out bf16 [N][K]
// ---------------------------------------------------------------------------
__global__ __launch_bounds__(256) void wconvT_kernel(const float* __restrict__ in,
                                                     __bf16* __restrict__ outT,
                                                     int K, int N) {
    __shared__ float tile[32][33];
    const int n0 = blockIdx.x * 32, k0 = blockIdx.y * 32;
    #pragma unroll
    for (int i = 0; i < 4; ++i) {
        int k = k0 + threadIdx.y + i * 8;
        tile[threadIdx.y + i * 8][threadIdx.x] = in[(size_t)k * N + n0 + threadIdx.x];
    }
    __syncthreads();
    #pragma unroll
    for (int i = 0; i < 4; ++i) {
        int n = n0 + threadIdx.y + i * 8;
        outT[(size_t)n * K + k0 + threadIdx.x] = (__bf16)tile[threadIdx.x][threadIdx.y + i * 8];
    }
}

// ---------------------------------------------------------------------------
// LayerNorm: one block per row of 1024; bf16 out (ws) or f32 out (d_out)
// ---------------------------------------------------------------------------
__global__ __launch_bounds__(256) void layernorm_kernel(const float* __restrict__ x,
                                                        const float* __restrict__ sc,
                                                        const float* __restrict__ bi,
                                                        __bf16* __restrict__ outb,
                                                        float* __restrict__ outf,
                                                        int f32out) {
    const int row = blockIdx.x, tid = threadIdx.x;
    const float4 xv = ((const float4*)(x + (size_t)row * DMODEL))[tid];
    float arr[4] = {xv.x, xv.y, xv.z, xv.w};
    float s = arr[0] + arr[1] + arr[2] + arr[3];
    float q = arr[0]*arr[0] + arr[1]*arr[1] + arr[2]*arr[2] + arr[3]*arr[3];
    __shared__ float rs[256], rq[256];
    rs[tid] = s; rq[tid] = q;
    __syncthreads();
    for (int off = 128; off > 0; off >>= 1) {
        if (tid < off) { rs[tid] += rs[tid + off]; rq[tid] += rq[tid + off]; }
        __syncthreads();
    }
    const float mu   = rs[0] * (1.0f / DMODEL);
    const float var  = rq[0] * (1.0f / DMODEL) - mu * mu;
    const float rstd = rsqrtf(var + LN_EPS);
    const int c0 = tid * 4;
    #pragma unroll
    for (int i = 0; i < 4; ++i) {
        float y = (arr[i] - mu) * rstd * sc[c0 + i] + bi[c0 + i];
        if (f32out) outf[(size_t)row * DMODEL + c0 + i] = y;
        else        outb[(size_t)row * DMODEL + c0 + i] = (__bf16)y;
    }
}

// ---------------------------------------------------------------------------
// GEMM: C = act(A[M,K](bf16) x W[K,N] + bias), W given transposed bf16 [N][K].
// Block tile 128x128, K-step 32, 8 waves (2x4), wave tile 64x32 (4x2 WMMA).
// Tiles DMA'd global->LDS by the Tensor Data Mover (wave 0 issues, TENSORcnt
// tracked, double-buffered).  Epilogue modes:
//   0: bf16 [M,N]            1: bf16 head layout [B*H][S][64]   (Q,K)
//   2: bf16 transposed heads [B*H][64][S]                       (V)
//   3: fp32 residual add into hres[M,N]                         (Wo, W2)
//   4: relu -> bf16 [M,N]                                       (W1)
// ---------------------------------------------------------------------------
__global__ __launch_bounds__(256) void gemm_kernel(const __bf16* __restrict__ A,
                                                   const __bf16* __restrict__ BT,
                                                   const float*  __restrict__ bias,
                                                   __bf16* __restrict__ outb,
                                                   float*  __restrict__ hres,
                                                   int M, int N, int K, int mode) {
    __shared__ __align__(16) __bf16 As[2][128][40];
    __shared__ __align__(16) __bf16 Bs[2][128][40];
    const int tid = threadIdx.x, lane = tid & 31, w = tid >> 5;
    const int m0 = blockIdx.x * 128, n0 = blockIdx.y * 128;
    const int wm = (w & 1) * 64, wn = (w >> 1) * 32;
    (void)M;

    v8f acc[4][2];
    #pragma unroll
    for (int mt = 0; mt < 4; ++mt)
        #pragma unroll
        for (int nt = 0; nt < 2; ++nt) acc[mt][nt] = (v8f){};

    const __bf16* Ag = A  + (size_t)m0 * K;
    const __bf16* Bg = BT + (size_t)n0 * K;

#if HAVE_TDM
    const unsigned asb[2] = {(unsigned)(size_t)&As[0][0][0], (unsigned)(size_t)&As[1][0][0]};
    const unsigned bsb[2] = {(unsigned)(size_t)&Bs[0][0][0], (unsigned)(size_t)&Bs[1][0][0]};
    if (w == 0) {                         // prime the pipeline (TDM ignores EXEC)
        tdm_load_tile(asb[0], Ag, K);
        tdm_load_tile(bsb[0], Bg, K);
    }
#endif

    int cur = 0;
    for (int k0 = 0; k0 < K; k0 += 32) {
#if HAVE_TDM
        if (w == 0) {
            if (k0 + 32 < K) {            // issue next tile pair, then wait
                tdm_load_tile(asb[cur ^ 1], Ag + k0 + 32, K);
                tdm_load_tile(bsb[cur ^ 1], Bg + k0 + 32, K);
                __builtin_amdgcn_s_wait_tensorcnt(2);   // current pair landed
            } else {
                __builtin_amdgcn_s_wait_tensorcnt(0);
            }
        }
        __syncthreads();                  // publish LDS tile to all 8 waves
#else
        {   // fallback: cooperative VGPR-staged copy (straight-line, no exec mask)
            const int r = tid >> 2, q = tid & 3;
            *(v8bf*)&As[0][r][q * 8]      = *(const v8bf*)(Ag + (size_t)r * K + k0 + q * 8);
            *(v8bf*)&Bs[0][r][q * 8]      = *(const v8bf*)(Bg + (size_t)r * K + k0 + q * 8);
            *(v8bf*)&As[0][r + 64][q * 8] = *(const v8bf*)(Ag + (size_t)(r + 64) * K + k0 + q * 8);
            *(v8bf*)&Bs[0][r + 64][q * 8] = *(const v8bf*)(Bg + (size_t)(r + 64) * K + k0 + q * 8);
        }
        __syncthreads();
#endif
        v16bf af[4], bfm[2];
        #pragma unroll
        for (int mt = 0; mt < 4; ++mt) af[mt]  = frag_a_ld(&As[cur][wm + mt * 16][0], 40, lane);
        #pragma unroll
        for (int nt = 0; nt < 2; ++nt) bfm[nt] = frag_b_ld(&Bs[cur][wn + nt * 16][0], 40, lane);
        #pragma unroll
        for (int mt = 0; mt < 4; ++mt)
            #pragma unroll
            for (int nt = 0; nt < 2; ++nt)
                acc[mt][nt] = wmma_bf16(af[mt], bfm[nt], acc[mt][nt]);
        __syncthreads();                  // tile fully consumed before reuse
#if HAVE_TDM
        cur ^= 1;
#endif
    }

    // Epilogue. C layout: VGPR v -> row v + 8*(lane/16); col = lane%16.
    const int h16 = lane >> 4, ln = lane & 15;
    #pragma unroll
    for (int mt = 0; mt < 4; ++mt) {
        #pragma unroll
        for (int nt = 0; nt < 2; ++nt) {
            #pragma unroll
            for (int v = 0; v < 8; ++v) {
                const int row = m0 + wm + mt * 16 + v + 8 * h16;
                const int col = n0 + wn + nt * 16 + ln;
                float val = acc[mt][nt][v] + bias[col];
                if (mode == 0) {
                    outb[(size_t)row * N + col] = (__bf16)val;
                } else if (mode == 1) {                 // [B*H][S][64]
                    const int b = row >> 11, s = row & (SEQ - 1);
                    const int hh = col >> 6, dl = col & (DK - 1);
                    outb[(((size_t)(b * NHEADS + hh)) * SEQ + s) * DK + dl] = (__bf16)val;
                } else if (mode == 2) {                 // [B*H][64][S]
                    const int b = row >> 11, s = row & (SEQ - 1);
                    const int hh = col >> 6, dl = col & (DK - 1);
                    outb[(((size_t)(b * NHEADS + hh)) * DK + dl) * SEQ + s] = (__bf16)val;
                } else if (mode == 3) {                 // residual accumulate
                    hres[(size_t)row * N + col] += val;
                } else {                                // relu -> bf16
                    outb[(size_t)row * N + col] = (__bf16)fmaxf(val, 0.0f);
                }
            }
        }
    }
}

// ---------------------------------------------------------------------------
// Flash attention: grid (SEQ/64, B*H), 128 threads = 4 waves, 16 q-rows/wave.
// Q,K: [B*H][S][64] bf16,  Vt: [B*H][64][S] bf16,  out: [B*S][D] bf16.
// ---------------------------------------------------------------------------
__global__ __launch_bounds__(128) void attention_kernel(const __bf16* __restrict__ Q,
                                                        const __bf16* __restrict__ Kc,
                                                        const __bf16* __restrict__ Vt,
                                                        const unsigned char* __restrict__ mask,
                                                        __bf16* __restrict__ outb) {
    __shared__ __align__(16) __bf16 Plds[4][16][40];
    const int lane = threadIdx.x & 31, w = threadIdx.x >> 5;
    const int h16 = lane >> 4, ln = lane & 15;
    const int bh = blockIdx.y, b = bh >> 4, hh = bh & (NHEADS - 1);
    const int q0 = blockIdx.x * 64 + w * 16;

    const __bf16* Qh = Q  + (size_t)bh * SEQ * DK;
    const __bf16* Kh = Kc + (size_t)bh * SEQ * DK;
    const __bf16* Vh = Vt + (size_t)bh * DK * SEQ;

    // Q fragments for this wave's 16 rows (K-dim 64 = two K=32 fragments)
    v16bf qf0 = frag_a_ld(Qh + (size_t)q0 * DK +  0, DK, lane);
    v16bf qf1 = frag_a_ld(Qh + (size_t)q0 * DK + 32, DK, lane);

    float m[8], l[8];
    v8f o[4];
    #pragma unroll
    for (int v = 0; v < 8; ++v) { m[v] = -1e30f; l[v] = 0.0f; }
    #pragma unroll
    for (int j = 0; j < 4; ++j) o[j] = (v8f){};

    for (int kk = 0; kk < SEQ; kk += 32) {
        if (kk + 32 < SEQ) {   // cache hint for next key block (global_prefetch_b8)
            __builtin_prefetch(Kh + (size_t)(kk + 32 + ln) * DK + h16 * 32, 0, 1);
            __builtin_prefetch(Vh + (size_t)(16 * h16 + ln) * SEQ + kk + 32, 0, 1);
        }
        // scores for two 16-key tiles: S = Q x K^T  (K rows are B^T layout)
        v16bf kf00 = frag_b_ld(Kh + (size_t)(kk     ) * DK +  0, DK, lane);
        v16bf kf01 = frag_b_ld(Kh + (size_t)(kk     ) * DK + 32, DK, lane);
        v16bf kf10 = frag_b_ld(Kh + (size_t)(kk + 16) * DK +  0, DK, lane);
        v16bf kf11 = frag_b_ld(Kh + (size_t)(kk + 16) * DK + 32, DK, lane);
        v8f s0 = (v8f){}, s1 = (v8f){};
        s0 = wmma_bf16(qf0, kf00, s0); s0 = wmma_bf16(qf1, kf01, s0);
        s1 = wmma_bf16(qf0, kf10, s1); s1 = wmma_bf16(qf1, kf11, s1);

        const unsigned char mk0 = mask[b * SEQ + kk + ln];
        const unsigned char mk1 = mask[b * SEQ + kk + 16 + ln];

        float p0a[8], p1a[8], alpha[8];
        #pragma unroll
        for (int v = 0; v < 8; ++v) {
            float a0 = mk0 ? s0[v] * QK_SCALE : -1e9f;
            float a1 = mk1 ? s1[v] * QK_SCALE : -1e9f;
            float t = fmaxf(a0, a1);
            t = fmaxf(t, __shfl_xor(t, 1, 32));
            t = fmaxf(t, __shfl_xor(t, 2, 32));
            t = fmaxf(t, __shfl_xor(t, 4, 32));
            t = fmaxf(t, __shfl_xor(t, 8, 32));
            const float mn = fmaxf(m[v], t);
            alpha[v] = exp2f((m[v] - mn) * LOG2E);
            const float p0 = exp2f((a0 - mn) * LOG2E);
            const float p1 = exp2f((a1 - mn) * LOG2E);
            float rsum = p0 + p1;
            rsum += __shfl_xor(rsum, 1, 32);
            rsum += __shfl_xor(rsum, 2, 32);
            rsum += __shfl_xor(rsum, 4, 32);
            rsum += __shfl_xor(rsum, 8, 32);
            l[v] = l[v] * alpha[v] + rsum;
            m[v] = mn;
            p0a[v] = p0; p1a[v] = p1;
        }
        #pragma unroll
        for (int j = 0; j < 4; ++j)
            #pragma unroll
            for (int v = 0; v < 8; ++v) o[j][v] *= alpha[v];

        // C-layout P -> LDS -> A-fragment (16 x 32 keys), bf16
        #pragma unroll
        for (int v = 0; v < 8; ++v) {
            const int r = v + 8 * h16;
            Plds[w][r][ln]      = (__bf16)p0a[v];
            Plds[w][r][16 + ln] = (__bf16)p1a[v];
        }
        asm volatile("s_wait_dscnt 0x0" ::: "memory");  // same-wave LDS RAW
        v16bf pf = frag_a_ld(&Plds[w][0][0], 40, lane);

        // O += P x V  (V^T rows give B^T layout, contiguous keys)
        #pragma unroll
        for (int j = 0; j < 4; ++j) {
            v16bf vf = frag_b_ld(Vh + (size_t)(16 * j) * SEQ + kk, SEQ, lane);
            o[j] = wmma_bf16(pf, vf, o[j]);
        }
    }

    // normalize and write output row-block: out[b*S+row][hh*64 + 16j + ln]
    #pragma unroll
    for (int j = 0; j < 4; ++j) {
        #pragma unroll
        for (int v = 0; v < 8; ++v) {
            const int row = q0 + v + 8 * h16;
            const float val = o[j][v] / l[v];
            outb[((size_t)(b * SEQ + row)) * DMODEL + hh * DK + 16 * j + ln] = (__bf16)val;
        }
    }
}

// ---------------------------------------------------------------------------
// Host orchestration
// ---------------------------------------------------------------------------
extern "C" void kernel_launch(void* const* d_in, const int* in_sizes, int n_in,
                              void* d_out, int out_size, void* d_ws, size_t ws_size,
                              hipStream_t stream) {
    (void)in_sizes; (void)n_in; (void)out_size; (void)ws_size;
    const float* x    = (const float*)d_in[0];
    const unsigned char* mask = (const unsigned char*)d_in[1];
    const float* wq = (const float*)d_in[2];  const float* bq = (const float*)d_in[3];
    const float* wk = (const float*)d_in[4];  const float* bk = (const float*)d_in[5];
    const float* wv = (const float*)d_in[6];  const float* bv = (const float*)d_in[7];
    const float* wo = (const float*)d_in[8];  const float* bo = (const float*)d_in[9];
    const float* w1 = (const float*)d_in[10]; const float* b1 = (const float*)d_in[11];
    const float* w2 = (const float*)d_in[12]; const float* b2 = (const float*)d_in[13];
    const float* ln1s = (const float*)d_in[14]; const float* ln1b = (const float*)d_in[15];
    const float* ln2s = (const float*)d_in[16]; const float* ln2b = (const float*)d_in[17];
    const float* lnfs = (const float*)d_in[18]; const float* lnfb = (const float*)d_in[19];

    char* ws = (char*)d_ws;
    const size_t MB = 1u << 20;
    __bf16* WqT  = (__bf16*)(ws + 0 * MB);    // [1024][1024]
    __bf16* WkT  = (__bf16*)(ws + 2 * MB);
    __bf16* WvT  = (__bf16*)(ws + 4 * MB);
    __bf16* WoT  = (__bf16*)(ws + 6 * MB);
    __bf16* W1T  = (__bf16*)(ws + 8 * MB);    // [4096][1024]
    __bf16* W2T  = (__bf16*)(ws + 16 * MB);   // [1024][4096]
    float*  hbuf = (float*)(ws + 24 * MB);    // residual stream [4096][1024] fp32
    __bf16* xn   = (__bf16*)(ws + 40 * MB);   // normalized activations bf16
    __bf16* att  = (__bf16*)(ws + 48 * MB);   // attention output bf16
    __bf16* Qb   = (__bf16*)(ws + 56 * MB);   // [B*H][S][64]
    __bf16* Kb   = (__bf16*)(ws + 64 * MB);
    __bf16* Vt   = (__bf16*)(ws + 72 * MB);   // [B*H][64][S]
    __bf16* ffn1 = (__bf16*)(ws + 56 * MB);   // aliases Q/K/V (dead by then), 32 MB

    // residual stream <- x
    hipMemcpyAsync(hbuf, x, (size_t)ROWS * DMODEL * sizeof(float),
                   hipMemcpyDeviceToDevice, stream);

    // one-time weight convert+transpose to bf16 [N][K] (weights shared across layers)
    dim3 tb(32, 8);
    wconvT_kernel<<<dim3(DMODEL / 32, DMODEL / 32), tb, 0, stream>>>(wq, WqT, DMODEL, DMODEL);
    wconvT_kernel<<<dim3(DMODEL / 32, DMODEL / 32), tb, 0, stream>>>(wk, WkT, DMODEL, DMODEL);
    wconvT_kernel<<<dim3(DMODEL / 32, DMODEL / 32), tb, 0, stream>>>(wv, WvT, DMODEL, DMODEL);
    wconvT_kernel<<<dim3(DMODEL / 32, DMODEL / 32), tb, 0, stream>>>(wo, WoT, DMODEL, DMODEL);
    wconvT_kernel<<<dim3(DFF / 32,    DMODEL / 32), tb, 0, stream>>>(w1, W1T, DMODEL, DFF);
    wconvT_kernel<<<dim3(DMODEL / 32, DFF / 32),    tb, 0, stream>>>(w2, W2T, DFF, DMODEL);

    const dim3 gD(ROWS / 128, DMODEL / 128);   // N = 1024
    const dim3 gF(ROWS / 128, DFF / 128);      // N = 4096
    const dim3 gA(SEQ / 64, BATCH * NHEADS);   // attention

    for (int layer = 0; layer < NLAYERS; ++layer) {
        // ---- attention sublayer (pre-LN) ----
        layernorm_kernel<<<ROWS, 256, 0, stream>>>(hbuf, ln1s, ln1b, xn, nullptr, 0);
        gemm_kernel<<<gD, 256, 0, stream>>>(xn, WqT, bq, Qb, nullptr, ROWS, DMODEL, DMODEL, 1);
        gemm_kernel<<<gD, 256, 0, stream>>>(xn, WkT, bk, Kb, nullptr, ROWS, DMODEL, DMODEL, 1);
        gemm_kernel<<<gD, 256, 0, stream>>>(xn, WvT, bv, Vt, nullptr, ROWS, DMODEL, DMODEL, 2);
        attention_kernel<<<gA, 128, 0, stream>>>(Qb, Kb, Vt, mask, att);
        gemm_kernel<<<gD, 256, 0, stream>>>(att, WoT, bo, nullptr, hbuf, ROWS, DMODEL, DMODEL, 3);
        // ---- FFN sublayer (pre-LN) ----
        layernorm_kernel<<<ROWS, 256, 0, stream>>>(hbuf, ln2s, ln2b, xn, nullptr, 0);
        gemm_kernel<<<gF, 256, 0, stream>>>(xn, W1T, b1, ffn1, nullptr, ROWS, DFF, DMODEL, 4);
        gemm_kernel<<<gD, 256, 0, stream>>>(ffn1, W2T, b2, nullptr, hbuf, ROWS, DMODEL, DFF, 3);
    }

    // final LayerNorm -> fp32 output
    layernorm_kernel<<<ROWS, 256, 0, stream>>>(hbuf, lnfs, lnfb, nullptr, (float*)d_out, 1);
}
